// RASGModel_3375844294927
// MI455X (gfx1250) — matrix operand order repaired and checked by reading, not compiled, not run on hardware
//
#include <hip/hip_runtime.h>
#include <hip/hip_bf16.h>

// ---------------------------------------------------------------------------
// RASG model (basis-decomposed R-GCN + attention pooling) for gfx1250.
// Dense GEMMs: v_wmma_f32_16x16x32_f16, A staged via LDS (ds_load_b128),
// B pre-packed into exact fragment order (one 32B load per lane per k-step).
// Edge gather reads contiguous 2KB f16 rows t[src][8*128]; scatter via f32
// atomics. Softmax pooling via atomic segment max/sum.
// ---------------------------------------------------------------------------

typedef __attribute__((ext_vector_type(16))) _Float16 v16h;
typedef __attribute__((ext_vector_type(8)))  _Float16 v8h;
typedef __attribute__((ext_vector_type(4)))  _Float16 v4h;
typedef __attribute__((ext_vector_type(8)))  float    v8f;

#define N_NODES   50000
#define N_EDGES   800000
#define NUM_GRAPHS 256
#define IN_FEAT   64
#define REL_DIM   32
#define HID       128
#define NUM_RELS  32
#define NUM_BASES 8
#define NUM_LAYERS 2
#define K_MAX     160   // largest GEMM K (attention: HID + REL_DIM)

// ------------------------- small elementwise kernels -----------------------

// Pack a row-major f32 B[K,Nc] into WMMA B-fragment order (f16):
// out[((nt*(K/32)+kc)*32 + lane)*16 + e], where for lane L:
//   n = nt*16 + (L&15),  k = kc*32 + (e>>3)*16 + (L>>4)*8 + (e&7)      (ISA 32x16 B layout)
__global__ void pack_B_k(const float* __restrict__ B, _Float16* __restrict__ out,
                         int K, int Nc) {
    const int total = K * Nc;
    int i = blockIdx.x * blockDim.x + threadIdx.x;
    if (i >= total) return;
    const int nkc = K >> 5;
    int e = i & 15;
    int L = (i >> 4) & 31;
    int rest = i >> 9;
    int kc = rest % nkc;
    int nt = rest / nkc;
    int n = nt * 16 + (L & 15);
    int k = kc * 32 + (e >> 3) * 16 + (L >> 4) * 8 + (e & 7);
    out[i] = (_Float16)B[(size_t)k * Nc + n];
}

// V[l][b][k][j] (f32) -> Vcat[l][k][b*128+j] (f32): row-major [128,1024] per
// layer so t = h @ Vcat is one GEMM and t rows are contiguous per node.
__global__ void pack_V32_k(const float* __restrict__ V, float* __restrict__ out, int count) {
    int i = blockIdx.x * blockDim.x + threadIdx.x;
    if (i >= count) return;
    int j = i & 127;
    int b = (i >> 7) & 7;
    int k = (i >> 10) & 127;
    int l = i >> 17;
    out[i] = V[(((size_t)l * NUM_BASES + b) * HID + k) * HID + j];
}

// x[n][0:64]=feat, x[n][64:96]=rel_emb[query_rel[n]] ; also er_f16[n][0:32]
__global__ void build_x_k(const float* __restrict__ feat, const int* __restrict__ qr,
                          const float* __restrict__ rel_emb,
                          _Float16* __restrict__ x, _Float16* __restrict__ er, int total) {
    int i = blockIdx.x * blockDim.x + threadIdx.x;
    if (i >= total) return;
    int n = i / 96, j = i - n * 96;
    float v;
    if (j < IN_FEAT) v = feat[(size_t)n * IN_FEAT + j];
    else {
        v = rel_emb[(size_t)qr[n] * REL_DIM + (j - IN_FEAT)];
        er[(size_t)n * REL_DIM + (j - IN_FEAT)] = (_Float16)v;
    }
    x[i] = (_Float16)v;
}

// xa[n] = [h_final(128), er(32)]
__global__ void build_xa_k(const _Float16* __restrict__ h, const _Float16* __restrict__ er,
                           _Float16* __restrict__ xa, int total) {
    int i = blockIdx.x * blockDim.x + threadIdx.x;
    if (i >= total) return;
    int n = i / 160, j = i - n * 160;
    xa[i] = (j < HID) ? h[(size_t)n * HID + j] : er[(size_t)n * REL_DIM + (j - HID)];
}

// ------------------------------- WMMA GEMM ---------------------------------
// D = act(A[M,K] @ B[K,Nc] + bias + addend).  A f16 row-major (LDS-staged);
// Bpack in fragment order (see pack_B_k).  One wave per 16x16 tile.
// M % 16 == 0, K % 32 == 0, Nc % 16 == 0.  act: 0=none, 1=relu, 2=tanh
__global__ void __launch_bounds__(256)
wmma_gemm_k(const _Float16* __restrict__ A, int lda,
            const _Float16* __restrict__ Bpack,
            const float* __restrict__ bias,
            const float* __restrict__ addend,
            float* __restrict__ Df32,
            _Float16* __restrict__ Df16,
            int ldd, int M, int Nc, int K, int act) {
    __shared__ _Float16 sA[16 * K_MAX];
    const int tid  = threadIdx.x;
    const int wave = tid >> 5;
    const int lane = tid & 31;
    const int m0 = blockIdx.x * 16;
    const int ntile = blockIdx.y * 8 + wave;
    const int nkc = K >> 5;

    // Cooperatively stage A tile [16][K] into LDS (shared by all 8 waves).
    for (int i = tid * 8; i < 16 * K; i += 256 * 8) {
        int r = i / K, k = i - r * K;
        *(v8h*)(&sA[r * K + k]) = *(const v8h*)(A + (size_t)(m0 + r) * lda + k);
    }
    __syncthreads();

    if (ntile * 16 >= Nc) return;
    const int n0 = ntile * 16;
    const int hs = lane >> 4;   // K half-select
    const int lm = lane & 15;   // row (A) / column (B) within tile

    v8f acc = {};
    const _Float16* sArow = sA + lm * K;
    const _Float16* Bp = Bpack + ((size_t)ntile * nkc * 32 + lane) * 16;
    for (int kc = 0; kc < nkc; ++kc) {
        // A fragment from LDS: elements 0..7 = K kc*32+8*hs.., 8..15 = +16.
        v8h lo = *(const v8h*)(sArow + kc * 32 + 8 * hs);
        v8h hi = *(const v8h*)(sArow + kc * 32 + 16 + 8 * hs);
        v16h a;
#pragma unroll
        for (int i = 0; i < 8; ++i) { a[i] = lo[i]; a[i + 8] = hi[i]; }
        // B fragment: one contiguous 32B load per lane (pre-packed).
        v16h b = *(const v16h*)(Bp + (size_t)kc * 512);
        acc = __builtin_amdgcn_wmma_f32_16x16x32_f16(
            /*neg_a=*/false, a, /*neg_b=*/false, b,
            /*c_mod=*/(short)0, acc, /*reuse_a=*/false, /*reuse_b=*/false);
    }

    const float bn = bias ? bias[n0 + lm] : 0.f;
#pragma unroll
    for (int r = 0; r < 8; ++r) {
        int m = m0 + r + 8 * hs;               // C/D layout: lanes16-31 -> M+8
        size_t off = (size_t)m * ldd + n0 + lm;
        float v = acc[r] + bn;
        if (addend) v += addend[off];
        if (act == 1) v = fmaxf(v, 0.f);
        else if (act == 2) v = tanhf(v);
        if (Df32) Df32[off] = v;
        if (Df16) Df16[off] = (_Float16)v;
    }
}

// ------------------------------ edge phase ---------------------------------
// One wave per edge (grid-stride): msg = sum_b comp[et][b] * t[src][b*128+:],
// atomically accumulated into agg[dst]. t row is one contiguous 2KB block.
__global__ void __launch_bounds__(256)
edge_msg_k(const _Float16* __restrict__ t, const float* __restrict__ comp_l,
           const int* __restrict__ src, const int* __restrict__ dst,
           const int* __restrict__ etypes, float* __restrict__ agg, int E) {
    const int lane = threadIdx.x & 31;
    const int wid = blockIdx.x * (blockDim.x >> 5) + (threadIdx.x >> 5);
    const int nw = gridDim.x * (blockDim.x >> 5);
    for (int e = wid; e < E; e += nw) {
        const int s = src[e], d = dst[e], et = etypes[e];
        const int en = e + nw;
        if (en < E)   // hide gather latency of next iteration (global_prefetch_b8)
            __builtin_prefetch(t + (size_t)src[en] * (NUM_BASES * HID), 0, 0);
        const float* c = comp_l + et * NUM_BASES;
        const _Float16* row = t + (size_t)s * (NUM_BASES * HID) + lane * 4;
        float a0 = 0.f, a1 = 0.f, a2 = 0.f, a3 = 0.f;
#pragma unroll
        for (int b = 0; b < NUM_BASES; ++b) {
            const float cb = c[b];
            v4h v = *(const v4h*)(row + b * HID);
            a0 += cb * (float)v[0];
            a1 += cb * (float)v[1];
            a2 += cb * (float)v[2];
            a3 += cb * (float)v[3];
        }
        float* ap = agg + (size_t)d * HID + lane * 4;
        atomicAdd(ap + 0, a0);
        atomicAdd(ap + 1, a1);
        atomicAdd(ap + 2, a2);
        atomicAdd(ap + 3, a3);
    }
}

// --------------------------- attention pooling -----------------------------

__global__ void __launch_bounds__(256)
score_k(const float* __restrict__ u, const float* __restrict__ wsc,
        const float* __restrict__ bsc, float* __restrict__ a, int N) {
    const int lane = threadIdx.x & 31;
    const int n = blockIdx.x * 8 + (threadIdx.x >> 5);
    if (n >= N) return;
    const float* row = u + (size_t)n * HID + lane * 4;
    const float* w = wsc + lane * 4;
    float s = row[0] * w[0] + row[1] * w[1] + row[2] * w[2] + row[3] * w[3];
    for (int o = 16; o > 0; o >>= 1) s += __shfl_xor(s, o, 32);
    if (lane == 0) a[n] = s + bsc[0];
}

__device__ __forceinline__ unsigned f32_key(float f) {
    unsigned b = __float_as_uint(f);
    return (b & 0x80000000u) ? ~b : (b | 0x80000000u);
}
__device__ __forceinline__ float key_f32(unsigned k) {
    return __uint_as_float((k & 0x80000000u) ? (k & 0x7FFFFFFFu) : ~k);
}

__global__ void segmax_k(const float* __restrict__ a, const int* __restrict__ gid,
                         unsigned* __restrict__ mk, int N) {
    int n = blockIdx.x * blockDim.x + threadIdx.x;
    if (n >= N) return;
    atomicMax(&mk[gid[n]], f32_key(a[n]));
}

__global__ void expsum_k(const float* __restrict__ a, const int* __restrict__ gid,
                         const unsigned* __restrict__ mk, float* __restrict__ ex,
                         float* __restrict__ ssum, int N) {
    int n = blockIdx.x * blockDim.x + threadIdx.x;
    if (n >= N) return;
    int g = gid[n];
    float e = __expf(a[n] - key_f32(mk[g]));
    ex[n] = e;
    atomicAdd(&ssum[g], e);
}

__global__ void __launch_bounds__(256)
pool_k(const float* __restrict__ h, const float* __restrict__ ex,
       const float* __restrict__ ssum, const int* __restrict__ gid,
       float* __restrict__ z, int N) {
    const int lane = threadIdx.x & 31;
    const int n = blockIdx.x * 8 + (threadIdx.x >> 5);
    if (n >= N) return;
    const int g = gid[n];
    const float alpha = ex[n] / ssum[g];
    const float* row = h + (size_t)n * HID + lane * 4;
    float* zp = z + (size_t)g * HID + lane * 4;
    atomicAdd(zp + 0, row[0] * alpha);
    atomicAdd(zp + 1, row[1] * alpha);
    atomicAdd(zp + 2, row[2] * alpha);
    atomicAdd(zp + 3, row[3] * alpha);
}

__global__ void __launch_bounds__(256)
out_k(const float* __restrict__ z, const float* __restrict__ wout,
      const float* __restrict__ bout, float* __restrict__ out, int G) {
    const int lane = threadIdx.x & 31;
    const int g = blockIdx.x * 8 + (threadIdx.x >> 5);
    if (g >= G) return;
    const float* row = z + (size_t)g * HID + lane * 4;
    const float* w = wout + lane * 4;
    float s = row[0] * w[0] + row[1] * w[1] + row[2] * w[2] + row[3] * w[3];
    for (int o = 16; o > 0; o >>= 1) s += __shfl_xor(s, o, 32);
    if (lane == 0) out[g] = s + bout[0];
}

// ------------------------------- launcher ----------------------------------

extern "C" void kernel_launch(void* const* d_in, const int* in_sizes, int n_in,
                              void* d_out, int out_size, void* d_ws, size_t ws_size,
                              hipStream_t stream) {
    (void)in_sizes; (void)n_in; (void)out_size; (void)ws_size;

    const float* feat     = (const float*)d_in[0];
    const int*   qrel     = (const int*)  d_in[1];
    const int*   src      = (const int*)  d_in[2];
    const int*   dst      = (const int*)  d_in[3];
    const int*   etypes   = (const int*)  d_in[4];
    const int*   graph_id = (const int*)  d_in[5];
    const float* rel_emb  = (const float*)d_in[6];
    const float* W_in     = (const float*)d_in[7];
    const float* b_in     = (const float*)d_in[8];
    const float* V        = (const float*)d_in[9];
    const float* comp     = (const float*)d_in[10];
    const float* W_loop   = (const float*)d_in[11];
    const float* b_loop   = (const float*)d_in[12];
    const float* W_attn   = (const float*)d_in[13];
    const float* b_attn   = (const float*)d_in[14];
    const float* w_score  = (const float*)d_in[15];
    const float* b_score  = (const float*)d_in[16];
    const float* w_out    = (const float*)d_in[17];
    const float* b_out    = (const float*)d_in[18];
    float* out = (float*)d_out;

    // ---- workspace bump allocator (256B aligned) ----
    size_t off = 0;
    char* ws = (char*)d_ws;
    auto alloc = [&](size_t bytes) -> void* {
        void* p = ws + off;
        off = (off + bytes + 255) & ~(size_t)255;
        return p;
    };
    _Float16* er_f16    = (_Float16*)alloc((size_t)N_NODES * REL_DIM * 2);
    _Float16* x_f16     = (_Float16*)alloc((size_t)N_NODES * 96 * 2);
    float*    Vcat_f32  = (float*)   alloc((size_t)NUM_LAYERS * HID * NUM_BASES * HID * 4);
    _Float16* Winp      = (_Float16*)alloc((size_t)96 * HID * 2);
    _Float16* Vp        = (_Float16*)alloc((size_t)NUM_LAYERS * HID * NUM_BASES * HID * 2);
    _Float16* Wloopp    = (_Float16*)alloc((size_t)NUM_LAYERS * HID * HID * 2);
    _Float16* Wattnp    = (_Float16*)alloc((size_t)160 * HID * 2);
    _Float16* hA_f16    = (_Float16*)alloc((size_t)N_NODES * HID * 2);
    _Float16* hB_f16    = (_Float16*)alloc((size_t)N_NODES * HID * 2);
    float*    h_f32     = (float*)   alloc((size_t)N_NODES * HID * 4);
    _Float16* t_f16     = (_Float16*)alloc((size_t)N_NODES * NUM_BASES * HID * 2);
    float*    agg       = (float*)   alloc((size_t)N_NODES * HID * 4);
    _Float16* xa_f16    = (_Float16*)alloc((size_t)N_NODES * 160 * 2);
    float*    u_f32     = (float*)   alloc((size_t)N_NODES * HID * 4);
    float*    a_f32     = (float*)   alloc((size_t)N_NODES * 4);
    float*    ex_f32    = (float*)   alloc((size_t)N_NODES * 4);
    unsigned* mkeys     = (unsigned*)alloc((size_t)NUM_GRAPHS * 4);
    float*    ssum      = (float*)   alloc((size_t)NUM_GRAPHS * 4);
    float*    z         = (float*)   alloc((size_t)NUM_GRAPHS * HID * 4);

    const int TB = 256;
    auto grid1 = [](int n, int tb) { return (n + tb - 1) / tb; };

    // ---- weight packing into WMMA fragment order ----
    pack_B_k<<<grid1(96 * HID, TB), TB, 0, stream>>>(W_in, Winp, 96, HID);
    for (int l = 0; l < NUM_LAYERS; ++l)
        pack_B_k<<<grid1(HID * HID, TB), TB, 0, stream>>>(
            W_loop + (size_t)l * HID * HID, Wloopp + (size_t)l * HID * HID, HID, HID);
    pack_B_k<<<grid1(160 * HID, TB), TB, 0, stream>>>(W_attn, Wattnp, 160, HID);
    pack_V32_k<<<grid1(NUM_LAYERS * HID * NUM_BASES * HID, TB), TB, 0, stream>>>(
        V, Vcat_f32, NUM_LAYERS * HID * NUM_BASES * HID);
    for (int l = 0; l < NUM_LAYERS; ++l)
        pack_B_k<<<grid1(HID * NUM_BASES * HID, TB), TB, 0, stream>>>(
            Vcat_f32 + (size_t)l * HID * NUM_BASES * HID,
            Vp + (size_t)l * HID * NUM_BASES * HID, HID, NUM_BASES * HID);
    build_x_k<<<grid1(N_NODES * 96, TB), TB, 0, stream>>>(
        feat, qrel, rel_emb, x_f16, er_f16, N_NODES * 96);

    const int MT = N_NODES / 16;   // 3125 row tiles (50000 % 16 == 0)

    // ---- h0 = relu([feat|er] @ W_in + b_in) ----
    wmma_gemm_k<<<dim3(MT, 1), TB, 0, stream>>>(
        x_f16, 96, Winp, b_in, nullptr,
        nullptr, hA_f16, HID, N_NODES, HID, 96, /*relu*/1);

    _Float16* hcur = hA_f16;
    _Float16* hnxt = hB_f16;

    for (int l = 0; l < NUM_LAYERS; ++l) {
        // t = h @ Vcat[l]  -> [N, 8*128] f16, contiguous per node
        wmma_gemm_k<<<dim3(MT, 8), TB, 0, stream>>>(
            hcur, HID, Vp + (size_t)l * HID * NUM_BASES * HID,
            nullptr, nullptr, nullptr, t_f16, NUM_BASES * HID,
            N_NODES, NUM_BASES * HID, HID, /*none*/0);
        // agg = 0 ; scatter-add weighted basis mixture per edge
        hipMemsetAsync(agg, 0, (size_t)N_NODES * HID * 4, stream);
        edge_msg_k<<<8192, TB, 0, stream>>>(
            t_f16, comp + (size_t)l * NUM_RELS * NUM_BASES, src, dst, etypes, agg, N_EDGES);
        // h = relu(agg + h @ W_loop[l] + b_loop[l])
        wmma_gemm_k<<<dim3(MT, 1), TB, 0, stream>>>(
            hcur, HID, Wloopp + (size_t)l * HID * HID,
            b_loop + (size_t)l * HID, agg,
            h_f32, hnxt, HID, N_NODES, HID, HID, /*relu*/1);
        _Float16* tmp = hcur; hcur = hnxt; hnxt = tmp;
    }

    // ---- attention scores: u = tanh([h|er] @ W_attn + b_attn) ----
    build_xa_k<<<grid1(N_NODES * 160, TB), TB, 0, stream>>>(
        hcur, er_f16, xa_f16, N_NODES * 160);
    wmma_gemm_k<<<dim3(MT, 1), TB, 0, stream>>>(
        xa_f16, 160, Wattnp, b_attn, nullptr,
        u_f32, nullptr, HID, N_NODES, HID, 160, /*tanh*/2);
    score_k<<<grid1(N_NODES, 8), TB, 0, stream>>>(u_f32, w_score, b_score, a_f32, N_NODES);

    // ---- segment softmax + weighted pooling ----
    hipMemsetAsync(mkeys, 0, (size_t)NUM_GRAPHS * 4, stream);
    hipMemsetAsync(ssum, 0, (size_t)NUM_GRAPHS * 4, stream);
    hipMemsetAsync(z, 0, (size_t)NUM_GRAPHS * HID * 4, stream);
    segmax_k<<<grid1(N_NODES, TB), TB, 0, stream>>>(a_f32, graph_id, mkeys, N_NODES);
    expsum_k<<<grid1(N_NODES, TB), TB, 0, stream>>>(a_f32, graph_id, mkeys, ex_f32, ssum, N_NODES);
    pool_k<<<grid1(N_NODES, 8), TB, 0, stream>>>(h_f32, ex_f32, ssum, graph_id, z, N_NODES);
    out_k<<<grid1(NUM_GRAPHS, 8), TB, 0, stream>>>(z, w_out, b_out, out, NUM_GRAPHS);
}
